// My_Net_66185446032028
// MI455X (gfx1250) — matrix operand order repaired and checked by reading, not compiled, not run on hardware
//
#include <hip/hip_runtime.h>
#include <cstdint>
#include <cstddef>

// ---------------------------------------------------------------------------
// Model constants (fixed by the reference)
// ---------------------------------------------------------------------------
#define H 256
#define N_DR 8000
#define N_P 16000
#define N_D 4000
#define N_MF 3000
#define N_BP 6000
#define N_CC 2000
#define N_PAIRS 32768
#define EPS 1e-5f

typedef _Float16 half_t;
typedef _Float16 h8   __attribute__((ext_vector_type(8)));
typedef _Float16 v16h __attribute__((ext_vector_type(16)));
typedef float    v8f  __attribute__((ext_vector_type(8)));

#define BM 32            // block rows
#define BN 128           // block cols
#define NT 4             // 16-wide N tiles per wave (wave computes 16x64)
#define LDSK 40          // 32 + 8 f16 pad (16B) -> bank-conflict-free, 16B aligned rows

enum { GEMM_STORE = 0, GEMM_RELU = 1, GEMM_ADD_RELU = 2 };

// Fused pair-gather descriptor: concat([src[0..2]][x_dr] , [src[3..6]][x_p]) along K.
// Each 32-wide K tile lies inside one 256-wide segment, so seg = k0 >> 8.
struct GatherA {
  const float* src[7];
  const int* xdr;
  const int* xp;
};

// ---------------------------------------------------------------------------
// WMMA GEMM: C(epilogue) <- A[M,K] @ B[K,N] (+ A2 @ B2) + bias
// 128 threads = 4 waves (2x2). Block tile 32x128; each wave does 16x64 =
// 4 WMMA tiles sharing one A fragment (A HBM traffic cut 4x vs 32-wide tiles).
// Requires K % 32 == 0 (true for all calls); M and N guarded.
// ---------------------------------------------------------------------------
template <int MODE, bool DUAL, bool GATHER>
__launch_bounds__(128)
__global__ void gemm_wmma_kernel(const float* __restrict__ A,  const float* __restrict__ B,
                                 const float* __restrict__ A2, const float* __restrict__ B2,
                                 const float* __restrict__ bias, float* __restrict__ C,
                                 int M, int N, int K, GatherA ga)
{
  __shared__ __align__(16) half_t As  [BM][LDSK];
  __shared__ __align__(16) half_t BsT [BN][LDSK];                 // [n][k]
  __shared__ __align__(16) half_t As2 [DUAL ? BM : 1][LDSK];
  __shared__ __align__(16) half_t Bs2T[DUAL ? BN : 1][LDSK];

  const int tid   = threadIdx.x;
  const int lane  = tid & 31;
  const int wave  = tid >> 5;
  const int waveM = wave >> 1, waveN = wave & 1;
  const int rowBase = blockIdx.y * BM;
  const int colBase = blockIdx.x * BN;

  // A loader: thread -> (row tid>>2, 8 contiguous k at (tid&3)*8)
  const int aRow = tid >> 2;
  const int aOff = (tid & 3) * 8;
  const int agr  = rowBase + aRow;
  // Gather row indices hoisted out of the K loop
  int gidx_dr = 0, gidx_p = 0;
  if (GATHER && agr < M) { gidx_dr = ga.xdr[agr]; gidx_p = ga.xp[agr]; }

  v8f acc[NT];
  #pragma unroll
  for (int t = 0; t < NT; ++t) acc[t] = (v8f){};

  for (int k0 = 0; k0 < K; k0 += 32) {
    __syncthreads();
    // ---- A tile: 2x global_load_b128 -> 1x ds_store_b128 ----
    {
      h8 av = (h8){};
      if (agr < M) {
        const float* srcA;
        if (GATHER) {
          const int seg = k0 >> 8, cc = k0 & 255;
          const int r = (seg < 3) ? gidx_dr : gidx_p;
          srcA = ga.src[seg] + (size_t)r * H + cc + aOff;
        } else {
          srcA = A + (size_t)agr * K + k0 + aOff;
          if (k0 + 32 < K) __builtin_prefetch(srcA + 32, 0, 1);
        }
        const float4 f0 = *(const float4*)(srcA + 0);
        const float4 f1 = *(const float4*)(srcA + 4);
        av[0] = (half_t)f0.x; av[1] = (half_t)f0.y; av[2] = (half_t)f0.z; av[3] = (half_t)f0.w;
        av[4] = (half_t)f1.x; av[5] = (half_t)f1.y; av[6] = (half_t)f1.z; av[7] = (half_t)f1.w;
      }
      *(h8*)&As[aRow][aOff] = av;
      if (DUAL) {
        h8 av2 = (h8){};
        if (agr < M) {
          const float* srcA2 = A2 + (size_t)agr * K + k0 + aOff;
          const float4 f0 = *(const float4*)(srcA2 + 0);
          const float4 f1 = *(const float4*)(srcA2 + 4);
          av2[0] = (half_t)f0.x; av2[1] = (half_t)f0.y; av2[2] = (half_t)f0.z; av2[3] = (half_t)f0.w;
          av2[4] = (half_t)f1.x; av2[5] = (half_t)f1.y; av2[6] = (half_t)f1.z; av2[7] = (half_t)f1.w;
        }
        *(h8*)&As2[aRow][aOff] = av2;
      }
    }
    // ---- B tile transposed in registers: coalesced b32 column reads,
    //      then 4x ds_store_b128 per thread into BsT[n][k] ----
    {
      const int gc = colBase + tid;            // one n-column per thread
      h8 bv[4]; h8 bv2[4];
      #pragma unroll
      for (int c = 0; c < 4; ++c) { bv[c] = (h8){}; bv2[c] = (h8){}; }
      if (gc < N) {
        const float* srcB = B + (size_t)k0 * N + gc;
        #pragma unroll
        for (int c = 0; c < 4; ++c)
          #pragma unroll
          for (int j = 0; j < 8; ++j)
            bv[c][j] = (half_t)srcB[(size_t)(c * 8 + j) * N];
        if (DUAL) {
          const float* srcB2 = B2 + (size_t)k0 * N + gc;
          #pragma unroll
          for (int c = 0; c < 4; ++c)
            #pragma unroll
            for (int j = 0; j < 8; ++j)
              bv2[c][j] = (half_t)srcB2[(size_t)(c * 8 + j) * N];
        }
      }
      #pragma unroll
      for (int c = 0; c < 4; ++c) *(h8*)&BsT[tid][c * 8] = bv[c];
      if (DUAL) {
        #pragma unroll
        for (int c = 0; c < 4; ++c) *(h8*)&Bs2T[tid][c * 8] = bv2[c];
      }
    }
    __syncthreads();

    // ISA lane layout (CDNA5 ISA): A 16x32 f16: lane half 0 holds K{0-7,16-23},
    // half 1 holds K{8-15,24-31}; B 32x16 f16: same K order as A.
    const int hf = lane >> 4;
    const int am = waveM * 16 + (lane & 15);
    v16h a, a2;
    {
      h8 alo = *(const h8*)&As[am][hf * 8];
      h8 ahi = *(const h8*)&As[am][16 + hf * 8];
      #pragma unroll
      for (int i = 0; i < 8; ++i) { a[i] = alo[i]; a[i + 8] = ahi[i]; }
    }
    if (DUAL) {
      h8 alo = *(const h8*)&As2[am][hf * 8];
      h8 ahi = *(const h8*)&As2[am][16 + hf * 8];
      #pragma unroll
      for (int i = 0; i < 8; ++i) { a2[i] = alo[i]; a2[i + 8] = ahi[i]; }
    }
    #pragma unroll
    for (int t = 0; t < NT; ++t) {
      const int bn = waveN * (NT * 16) + t * 16 + (lane & 15);
      {
        h8 blo = *(const h8*)&BsT[bn][hf * 8];
        h8 bhi = *(const h8*)&BsT[bn][16 + hf * 8];
        v16h b;
        #pragma unroll
        for (int i = 0; i < 8; ++i) { b[i] = blo[i]; b[i + 8] = bhi[i]; }
        acc[t] = __builtin_amdgcn_wmma_f32_16x16x32_f16(false, a, false, b, (short)0, acc[t],
                                                        false, false);
      }
      if (DUAL) {
        h8 blo = *(const h8*)&Bs2T[bn][hf * 8];
        h8 bhi = *(const h8*)&Bs2T[bn][16 + hf * 8];
        v16h b;
        #pragma unroll
        for (int i = 0; i < 8; ++i) { b[i] = blo[i]; b[i + 8] = bhi[i]; }
        acc[t] = __builtin_amdgcn_wmma_f32_16x16x32_f16(false, a2, false, b, (short)0, acc[t],
                                                        false, false);
      }
    }
  }

  // ---- epilogue (branch-free per template MODE) ----
  // C/D layout: VGPR r -> M = r + 8*(lane>=16), N = lane&15
  const int hf   = lane >> 4;
  const int row0 = rowBase + waveM * 16 + hf * 8;
  #pragma unroll
  for (int t = 0; t < NT; ++t) {
    const int col = colBase + waveN * (NT * 16) + t * 16 + (lane & 15);
    if (col < N) {
      const float bv = bias ? bias[col] : 0.0f;
      #pragma unroll
      for (int r = 0; r < 8; ++r) {
        const int row = row0 + r;
        if (row < M) {
          const float v = acc[t][r] + bv;
          const size_t o = (size_t)row * N + col;
          if      (MODE == GEMM_ADD_RELU) C[o] += fmaxf(v, 0.0f);
          else if (MODE == GEMM_RELU)     C[o]  = fmaxf(v, 0.0f);
          else                            C[o]  = v;
        }
      }
    }
  }
}

// ---------------------------------------------------------------------------
// Elementwise / graph kernels
// ---------------------------------------------------------------------------
__global__ void zero_kernel(float* __restrict__ p, long long n) {
  for (long long i = blockIdx.x * (long long)blockDim.x + threadIdx.x; i < n;
       i += (long long)gridDim.x * blockDim.x) p[i] = 0.0f;
}

__global__ void bias_relu_kernel(const float* __restrict__ W, const float* __restrict__ b,
                                 float* __restrict__ o, long long n, int cols) {
  for (long long i = blockIdx.x * (long long)blockDim.x + threadIdx.x; i < n;
       i += (long long)gridDim.x * blockDim.x)
    o[i] = fmaxf(W[i] + b[(int)(i % cols)], 0.0f);
}

// segment_sum over edges: dst[e1] += src[e0]   (e = [2, ne] int32)
__global__ void scatter_add_kernel(const float* __restrict__ src, const int* __restrict__ e,
                                   long long ne, float* __restrict__ dst) {
  const int CH = H / 4;
  const long long total = ne * CH;
  for (long long i = blockIdx.x * (long long)blockDim.x + threadIdx.x; i < total;
       i += (long long)gridDim.x * blockDim.x) {
    const long long ei = i / CH;
    const int c = (int)(i % CH) * 4;
    const int s = e[ei], d = e[ne + ei];
    const float4 v = *(const float4*)(src + (size_t)s * H + c);
    float* p = dst + (size_t)d * H + c;
    atomicAdd(p + 0, v.x); atomicAdd(p + 1, v.y);
    atomicAdd(p + 2, v.z); atomicAdd(p + 3, v.w);
  }
}

// segment_max of non-negative (post-ReLU) values; dst pre-zeroed, so u32-bit-pattern
// max equals float max and empty segments stay exactly 0.0f (matches isfinite fixup).
__global__ void scatter_max_kernel(const float* __restrict__ src, const int* __restrict__ e,
                                   long long ne, float* __restrict__ dst) {
  const int CH = H / 4;
  const long long total = ne * CH;
  for (long long i = blockIdx.x * (long long)blockDim.x + threadIdx.x; i < total;
       i += (long long)gridDim.x * blockDim.x) {
    const long long ei = i / CH;
    const int c = (int)(i % CH) * 4;
    const int s = e[ei], d = e[ne + ei];
    const float4 v = *(const float4*)(src + (size_t)s * H + c);
    unsigned int* p = (unsigned int*)(dst + (size_t)d * H + c);
    atomicMax(p + 0, __float_as_uint(v.x)); atomicMax(p + 1, __float_as_uint(v.y));
    atomicMax(p + 2, __float_as_uint(v.z)); atomicMax(p + 3, __float_as_uint(v.w));
  }
}

__global__ void bn_stats_kernel(const float* __restrict__ x, int M, int N,
                                float* __restrict__ mu, float* __restrict__ rs) {
  __shared__ float ss[256], sq[256];
  const int c = blockIdx.x;
  float s = 0.0f, q = 0.0f;
  for (int r = threadIdx.x; r < M; r += blockDim.x) {
    const float v = x[(size_t)r * N + c];
    s += v; q += v * v;
  }
  ss[threadIdx.x] = s; sq[threadIdx.x] = q;
  __syncthreads();
  for (int st = 128; st > 0; st >>= 1) {
    if (threadIdx.x < st) { ss[threadIdx.x] += ss[threadIdx.x + st]; sq[threadIdx.x] += sq[threadIdx.x + st]; }
    __syncthreads();
  }
  if (threadIdx.x == 0) {
    const float m   = ss[0] / (float)M;
    const float var = sq[0] / (float)M - m * m;
    mu[c] = m;
    rs[c] = rsqrtf(var + EPS);
  }
}

__global__ void bn_apply_relu_kernel(float* __restrict__ x, long long n, int N,
                                     const float* __restrict__ mu, const float* __restrict__ rs,
                                     const float* __restrict__ g,  const float* __restrict__ be) {
  for (long long i = blockIdx.x * (long long)blockDim.x + threadIdx.x; i < n;
       i += (long long)gridDim.x * blockDim.x) {
    const int c = (int)(i % N);
    const float v = g[c] * (x[i] - mu[c]) * rs[c] + be[c];
    x[i] = fmaxf(v, 0.0f);
  }
}

__global__ void head_kernel(const float* __restrict__ z, const float* __restrict__ Wo,
                            const float* __restrict__ bo, float* __restrict__ out, int M) {
  const int r = blockIdx.x * blockDim.x + threadIdx.x;
  if (r >= M) return;
  float s = bo[0];
  #pragma unroll
  for (int k = 0; k < 64; ++k) s += z[(size_t)r * 64 + k] * Wo[k];
  out[r] = 1.0f / (1.0f + expf(-s));
}

// ---------------------------------------------------------------------------
// Host-side GEMM dispatch over template instantiations
// ---------------------------------------------------------------------------
static void launch_gemm(const float* A, const float* B, const float* A2, const float* B2,
                        const float* bias, float* C, int M, int N, int K, int mode,
                        hipStream_t s)
{
  dim3 g((N + BN - 1) / BN, (M + BM - 1) / BM);
  GatherA ga = {};
  if (A2) {
    gemm_wmma_kernel<GEMM_ADD_RELU, true, false><<<g, 128, 0, s>>>(A, B, A2, B2, bias, C, M, N, K, ga);
  } else if (mode == GEMM_ADD_RELU) {
    gemm_wmma_kernel<GEMM_ADD_RELU, false, false><<<g, 128, 0, s>>>(A, B, nullptr, nullptr, bias, C, M, N, K, ga);
  } else if (mode == GEMM_RELU) {
    gemm_wmma_kernel<GEMM_RELU, false, false><<<g, 128, 0, s>>>(A, B, nullptr, nullptr, bias, C, M, N, K, ga);
  } else {
    gemm_wmma_kernel<GEMM_STORE, false, false><<<g, 128, 0, s>>>(A, B, nullptr, nullptr, bias, C, M, N, K, ga);
  }
}

static void launch_gemm_gather(const GatherA& ga, const float* B, const float* bias, float* C,
                               int M, int N, int K, hipStream_t s)
{
  dim3 g((N + BN - 1) / BN, (M + BM - 1) / BM);
  gemm_wmma_kernel<GEMM_STORE, false, true><<<g, 128, 0, s>>>(nullptr, B, nullptr, nullptr,
                                                              bias, C, M, N, K, ga);
}

// ---------------------------------------------------------------------------
// Host orchestration
// ---------------------------------------------------------------------------
extern "C" void kernel_launch(void* const* d_in, const int* in_sizes, int n_in,
                              void* d_out, int out_size, void* d_ws, size_t ws_size,
                              hipStream_t stream)
{
  const float* finger   = (const float*)d_in[0];
  const float* seq      = (const float*)d_in[1];
  // d_in[2..5] are identity one-hot features: embedding reduces to relu(W + b).
  const float* W_dr_emb = (const float*)d_in[6];  const float* b_dr_emb = (const float*)d_in[7];
  const float* W_p_emb  = (const float*)d_in[8];  const float* b_p_emb  = (const float*)d_in[9];
  const float* W_d_emb  = (const float*)d_in[10]; const float* b_d_emb  = (const float*)d_in[11];
  const float* W_mf_emb = (const float*)d_in[12]; const float* b_mf_emb = (const float*)d_in[13];
  const float* W_bp_emb = (const float*)d_in[14]; const float* b_bp_emb = (const float*)d_in[15];
  const float* W_cc_emb = (const float*)d_in[16]; const float* b_cc_emb = (const float*)d_in[17];
  const float* W_gosim  = (const float*)d_in[18]; const float* b_gosim  = (const float*)d_in[19];
  const float* W_go2p   = (const float*)d_in[20]; const float* b_go2p   = (const float*)d_in[21];
  const float* Wpool    = (const float*)d_in[22]; const float* bpool    = (const float*)d_in[23];
  const float* Wself    = (const float*)d_in[24]; const float* Wneigh   = (const float*)d_in[25];
  const float* bsage    = (const float*)d_in[26];
  const float* W1 = (const float*)d_in[27]; const float* b1 = (const float*)d_in[28];
  const float* g1 = (const float*)d_in[29]; const float* be1 = (const float*)d_in[30];
  const float* W2 = (const float*)d_in[31]; const float* b2 = (const float*)d_in[32];
  const float* g2 = (const float*)d_in[33]; const float* be2 = (const float*)d_in[34];
  const float* W3 = (const float*)d_in[35]; const float* b3 = (const float*)d_in[36];
  const float* g3 = (const float*)d_in[37]; const float* be3 = (const float*)d_in[38];
  const float* Wo = (const float*)d_in[39]; const float* bo = (const float*)d_in[40];
  const int* x_dr = (const int*)d_in[41];
  const int* x_p  = (const int*)d_in[42];

  const int* E[14];
  long long NE[14];
  for (int i = 0; i < 14; ++i) { E[i] = (const int*)d_in[43 + i]; NE[i] = in_sizes[43 + i] / 2; }
  // 0:mf_sim 1:bp_sim 2:cc_sim 3:mf_p 4:bp_p 5:cc_p 6:dtdr 7:dmdr 8:dp 9:drtd 10:drmd 11:pd 12:ddi 13:ppi

  char* wsb = (char*)d_ws;
  size_t off = 0;
  auto alloc = [&](size_t elems) -> float* {
    float* p = (float*)(wsb + off);
    off += (elems * sizeof(float) + 255) & ~(size_t)255;
    return p;
  };
  float* h_dr_f = alloc((size_t)N_DR * H);
  float* h_p_s  = alloc((size_t)N_P  * H);
  float* h_d    = alloc((size_t)N_D  * H);
  float* h_mf   = alloc((size_t)N_MF * H);
  float* h_bp   = alloc((size_t)N_BP * H);
  float* h_cc   = alloc((size_t)N_CC * H);
  float* go_mf  = alloc((size_t)N_MF * H);
  float* go_bp  = alloc((size_t)N_BP * H);
  float* go_cc  = alloc((size_t)N_CC * H);
  float* h_p_GO = alloc((size_t)N_P  * H);
  float* agg    = alloc((size_t)N_P  * H);   // scatter-add scratch (max dst = N_P)
  float* tmp_m  = alloc((size_t)N_P  * H);   // SAGE pool output (max src = N_P)
  float* neigh  = alloc((size_t)N_P  * H);   // SAGE segment-max scratch
  float* h_dr1  = alloc((size_t)N_DR * H);
  float* h_p1   = alloc((size_t)N_P  * H);
  float* h_d1   = alloc((size_t)N_D  * H);
  float* h_dr2  = alloc((size_t)N_DR * H);
  float* h_p2   = alloc((size_t)N_P  * H);
  float* z1     = alloc((size_t)N_PAIRS * 2 * H);
  float* z2     = alloc((size_t)N_PAIRS * H);
  float* z3     = alloc((size_t)N_PAIRS * 64);
  float* mu     = alloc(2 * H);
  float* rsg    = alloc(2 * H);
  (void)ws_size; (void)n_in;

  auto zero = [&](float* p, long long n) { zero_kernel<<<1024, 256, 0, stream>>>(p, n); };
  auto gemm = [&](const float* A, const float* B, const float* A2, const float* B2,
                  const float* bias, float* C, int M, int N, int K, int mode) {
    launch_gemm(A, B, A2, B2, bias, C, M, N, K, mode, stream);
  };
  auto scat_add = [&](const float* src, const int* e, long long ne, float* dst) {
    scatter_add_kernel<<<4096, 256, 0, stream>>>(src, e, ne, dst);
  };
  auto scat_max = [&](const float* src, const int* e, long long ne, float* dst) {
    scatter_max_kernel<<<4096, 256, 0, stream>>>(src, e, ne, dst);
  };
  // GCN relation: outacc += relu(segment_sum(hsrc[e0] -> e1) @ W + b)
  auto gcn_into = [&](const float* hsrc, const int* e, long long ne, int ndst,
                      const float* W, const float* b, float* outacc) {
    zero(agg, (long long)ndst * H);
    scat_add(hsrc, e, ne, agg);
    gemm(agg, W, nullptr, nullptr, b, outacc, ndst, H, H, GEMM_ADD_RELU);
  };
  // SAGE-pool relation i: accbuf += relu(hd@Wself_i + segmax(relu(hs@Wp_i+bp_i))@Wneigh_i + bsage_i)
  auto sage_into = [&](int i, const float* hs, int ns, const float* hd, int nd,
                       const int* e, long long ne, float* accbuf) {
    gemm(hs, Wpool + (size_t)i * H * H, nullptr, nullptr, bpool + (size_t)i * H,
         tmp_m, ns, H, H, GEMM_RELU);
    zero(neigh, (long long)nd * H);
    scat_max(tmp_m, e, ne, neigh);
    gemm(hd, Wself + (size_t)i * H * H, neigh, Wneigh + (size_t)i * H * H,
         bsage + (size_t)i * H, accbuf, nd, H, H, GEMM_ADD_RELU);
  };

  // ---- Embeddings ----
  gemm(finger, W_dr_emb, nullptr, nullptr, b_dr_emb, h_dr_f, N_DR, H, 1024, GEMM_RELU);
  gemm(seq,    W_p_emb,  nullptr, nullptr, b_p_emb,  h_p_s,  N_P,  H, 1024, GEMM_RELU);
  bias_relu_kernel<<<2048, 256, 0, stream>>>(W_d_emb,  b_d_emb,  h_d,  (long long)N_D  * H, H);
  bias_relu_kernel<<<2048, 256, 0, stream>>>(W_mf_emb, b_mf_emb, h_mf, (long long)N_MF * H, H);
  bias_relu_kernel<<<2048, 256, 0, stream>>>(W_bp_emb, b_bp_emb, h_bp, (long long)N_BP * H, H);
  bias_relu_kernel<<<2048, 256, 0, stream>>>(W_cc_emb, b_cc_emb, h_cc, (long long)N_CC * H, H);

  // ---- GO sim GCN + residual:  go_x = h_x + relu(gcn) ----
  hipMemcpyAsync(go_mf, h_mf, (size_t)N_MF * H * sizeof(float), hipMemcpyDeviceToDevice, stream);
  hipMemcpyAsync(go_bp, h_bp, (size_t)N_BP * H * sizeof(float), hipMemcpyDeviceToDevice, stream);
  hipMemcpyAsync(go_cc, h_cc, (size_t)N_CC * H * sizeof(float), hipMemcpyDeviceToDevice, stream);
  gcn_into(h_mf, E[0], NE[0], N_MF, W_gosim + 0 * H * H, b_gosim + 0 * H, go_mf);
  gcn_into(h_bp, E[1], NE[1], N_BP, W_gosim + 1 * H * H, b_gosim + 1 * H, go_bp);
  gcn_into(h_cc, E[2], NE[2], N_CC, W_gosim + 2 * H * H, b_gosim + 2 * H, go_cc);

  // ---- GO -> protein (hetero sum of three GCNs) ----
  zero(h_p_GO, (long long)N_P * H);
  gcn_into(go_mf, E[3], NE[3], N_P, W_go2p + 0 * H * H, b_go2p + 0 * H, h_p_GO);
  gcn_into(go_bp, E[4], NE[4], N_P, W_go2p + 1 * H * H, b_go2p + 1 * H, h_p_GO);
  gcn_into(go_cc, E[5], NE[5], N_P, W_go2p + 2 * H * H, b_go2p + 2 * H, h_p_GO);

  // ---- Hetero SAGE layer 1 ----
  zero(h_dr1, (long long)N_DR * H);
  zero(h_p1,  (long long)N_P  * H);
  zero(h_d1,  (long long)N_D  * H);
  sage_into(0, h_d,    N_D,  h_dr_f, N_DR, E[6],  NE[6],  h_dr1);
  sage_into(1, h_d,    N_D,  h_dr_f, N_DR, E[7],  NE[7],  h_dr1);
  sage_into(6, h_dr_f, N_DR, h_dr_f, N_DR, E[12], NE[12], h_dr1);
  sage_into(2, h_d,    N_D,  h_p_s,  N_P,  E[8],  NE[8],  h_p1);
  sage_into(7, h_p_s,  N_P,  h_p_s,  N_P,  E[13], NE[13], h_p1);
  sage_into(3, h_dr_f, N_DR, h_d,    N_D,  E[9],  NE[9],  h_d1);
  sage_into(4, h_dr_f, N_DR, h_d,    N_D,  E[10], NE[10], h_d1);
  sage_into(5, h_p_s,  N_P,  h_d,    N_D,  E[11], NE[11], h_d1);

  // ---- Hetero SAGE layer 2 (d-output unused downstream, skip it) ----
  zero(h_dr2, (long long)N_DR * H);
  zero(h_p2,  (long long)N_P  * H);
  sage_into(0, h_d1,  N_D,  h_dr1, N_DR, E[6],  NE[6],  h_dr2);
  sage_into(1, h_d1,  N_D,  h_dr1, N_DR, E[7],  NE[7],  h_dr2);
  sage_into(6, h_dr1, N_DR, h_dr1, N_DR, E[12], NE[12], h_dr2);
  sage_into(2, h_d1,  N_D,  h_p1,  N_P,  E[8],  NE[8],  h_p2);
  sage_into(7, h_p1,  N_P,  h_p1,  N_P,  E[13], NE[13], h_p2);

  // ---- MLP head: pair-gather fused into the first GEMM's A loads ----
  GatherA ga;
  ga.src[0] = h_dr_f; ga.src[1] = h_dr1; ga.src[2] = h_dr2;
  ga.src[3] = h_p_s;  ga.src[4] = h_p1;  ga.src[5] = h_p2; ga.src[6] = h_p_GO;
  ga.xdr = x_dr; ga.xp = x_p;
  launch_gemm_gather(ga, W1, b1, z1, N_PAIRS, 2 * H, 7 * H, stream);
  bn_stats_kernel<<<2 * H, 256, 0, stream>>>(z1, N_PAIRS, 2 * H, mu, rsg);
  bn_apply_relu_kernel<<<4096, 256, 0, stream>>>(z1, (long long)N_PAIRS * 2 * H, 2 * H, mu, rsg, g1, be1);

  gemm(z1, W2, nullptr, nullptr, b2, z2, N_PAIRS, H, 2 * H, GEMM_STORE);
  bn_stats_kernel<<<H, 256, 0, stream>>>(z2, N_PAIRS, H, mu, rsg);
  bn_apply_relu_kernel<<<4096, 256, 0, stream>>>(z2, (long long)N_PAIRS * H, H, mu, rsg, g2, be2);

  gemm(z2, W3, nullptr, nullptr, b3, z3, N_PAIRS, 64, H, GEMM_STORE);
  bn_stats_kernel<<<64, 256, 0, stream>>>(z3, N_PAIRS, 64, mu, rsg);
  bn_apply_relu_kernel<<<2048, 256, 0, stream>>>(z3, (long long)N_PAIRS * 64, 64, mu, rsg, g3, be3);

  head_kernel<<<(N_PAIRS + 255) / 256, 256, 0, stream>>>(z3, Wo, bo, (float*)d_out, N_PAIRS);
  (void)out_size;
}
